// GraphEmbedding_76639396429912
// MI455X (gfx1250) — compile-verified, matrix operand order checked by reading, not run on hardware
//
#include <hip/hip_runtime.h>
#include <hip/hip_bf16.h>

#define N_NODES  8192
#define D_DIM    384
#define N_EDGES  262144
#define N_REL    4
#define N_LAYERS 2
#define BATCH    64
#define ND       ((size_t)N_NODES * D_DIM)

typedef __attribute__((ext_vector_type(16))) _Float16 v16h;
typedef __attribute__((ext_vector_type(8)))  _Float16 v8h;
typedef __attribute__((ext_vector_type(8)))  float    v8f;

// ---------------------------------------------------------------------------
// BatchNorm statistics: one block per feature column
// ---------------------------------------------------------------------------
__global__ __launch_bounds__(256) void bn_stats_kernel(const float* __restrict__ x,
                                                       float* __restrict__ mu,
                                                       float* __restrict__ rs) {
    const int d = blockIdx.x;
    const int tid = threadIdx.x;
    float s = 0.f, s2 = 0.f;
    for (int n = tid; n < N_NODES; n += 256) {
        float v = x[(size_t)n * D_DIM + d];
        s += v; s2 += v * v;
    }
    __shared__ float sh[256], sh2[256];
    sh[tid] = s; sh2[tid] = s2;
    __syncthreads();
    for (int st = 128; st > 0; st >>= 1) {
        if (tid < st) { sh[tid] += sh[tid + st]; sh2[tid] += sh2[tid + st]; }
        __syncthreads();
    }
    if (tid == 0) {
        float m = sh[0] * (1.0f / N_NODES);
        float var = sh2[0] * (1.0f / N_NODES) - m * m;
        mu[d] = m;
        rs[d] = rsqrtf(var + 1e-5f);
    }
}

// ---------------------------------------------------------------------------
// Apply BN, emit h (f32), h16 (f16), per-row inverse L2 norm
// ---------------------------------------------------------------------------
__global__ __launch_bounds__(128) void bn_apply_kernel(const float* __restrict__ x,
                                                       const float* __restrict__ g,
                                                       const float* __restrict__ b,
                                                       const float* __restrict__ mu,
                                                       const float* __restrict__ rs,
                                                       float* __restrict__ h,
                                                       _Float16* __restrict__ h16,
                                                       float* __restrict__ rnorm) {
    const int n = blockIdx.x;
    const int tid = threadIdx.x;
    float ss = 0.f;
#pragma unroll
    for (int j = 0; j < 3; ++j) {
        int d = tid + j * 128;
        float v = g[d] * (x[(size_t)n * D_DIM + d] - mu[d]) * rs[d] + b[d];
        h[(size_t)n * D_DIM + d] = v;
        h16[(size_t)n * D_DIM + d] = (_Float16)v;
        ss += v * v;
    }
    __shared__ float sh[128];
    sh[tid] = ss;
    __syncthreads();
    for (int st = 64; st > 0; st >>= 1) {
        if (tid < st) sh[tid] += sh[tid + st];
        __syncthreads();
    }
    if (tid == 0) rnorm[n] = rsqrtf(sh[0]);
}

// ---------------------------------------------------------------------------
// Per-edge cosine weight: one wave32 per edge
// ---------------------------------------------------------------------------
__global__ __launch_bounds__(256) void edge_weight_kernel(const float* __restrict__ h,
                                                          const float* __restrict__ rnorm,
                                                          const int* __restrict__ ei,
                                                          float* __restrict__ w_e) {
    const int e = blockIdx.x * 8 + (threadIdx.x >> 5);
    const int lane = threadIdx.x & 31;
    const int src = ei[e];
    const int dst = ei[N_EDGES + e];
    const float* hs = h + (size_t)src * D_DIM;
    const float* hd = h + (size_t)dst * D_DIM;
    float s = 0.f;
#pragma unroll
    for (int j = 0; j < 12; ++j) {
        int d = lane + j * 32;
        s += hs[d] * hd[d];
    }
    for (int m = 16; m > 0; m >>= 1) s += __shfl_xor(s, m, 32);
    if (lane == 0) w_e[e] = s * rnorm[src] * rnorm[dst];
}

// ---------------------------------------------------------------------------
// fp32 -> f16 conversion, zero fill, reciprocal-clamp helpers
// ---------------------------------------------------------------------------
__global__ __launch_bounds__(256) void cvt_f32_f16_kernel(const float* __restrict__ s,
                                                          _Float16* __restrict__ d, int n) {
    int i = blockIdx.x * 256 + threadIdx.x;
    if (i < n) d[i] = (_Float16)s[i];
}

__global__ __launch_bounds__(256) void fill_zero_kernel(float* __restrict__ p, int n) {
    int i = blockIdx.x * 256 + threadIdx.x;
    if (i < n) p[i] = 0.0f;
}

__global__ __launch_bounds__(256) void inv_clamp_kernel(float* __restrict__ p, int n) {
    int i = blockIdx.x * 256 + threadIdx.x;
    if (i < n) p[i] = 1.0f / fmaxf(p[i], 1.0f);
}

// ---------------------------------------------------------------------------
// Degree counts: per-relation + total, via f32 atomics
// ---------------------------------------------------------------------------
__global__ __launch_bounds__(256) void count_kernel(const int* __restrict__ ei,
                                                    const int* __restrict__ et,
                                                    float* __restrict__ cnt_rel,
                                                    float* __restrict__ cnt_all) {
    int e = blockIdx.x * 256 + threadIdx.x;
    if (e >= N_EDGES) return;
    int dst = ei[N_EDGES + e];
    int r = et[e];
    atomicAdd(cnt_rel + (size_t)r * N_NODES + dst, 1.0f);
    atomicAdd(cnt_all + dst, 1.0f);
}

// ---------------------------------------------------------------------------
// WMMA f16 GEMM:  C[m][n] = sum_k A16[m][k] * W16[n][k] + bias[n]
// wave computes a 16(M) x 64(N) tile; block = 4 waves = 64x64 tile.
// A/B lane layouts match the ISA 16-bit A (16x32) / B (32x16) VGPR maps:
//   lane L: row/col = L&15, K chunks at (L>>4)*8 and (L>>4)*8 + 16.
// ---------------------------------------------------------------------------
__global__ __launch_bounds__(128) void gemm_f16_wmma_kernel(
        const _Float16* __restrict__ A,
        const _Float16* __restrict__ W,
        const float* __restrict__ bias,
        float* __restrict__ C,
        long wStrideZ, long biasStrideZ, long cStrideZ) {
    const int lane = threadIdx.x & 31;
    const int wave = threadIdx.x >> 5;
    const int z = blockIdx.z;
    const _Float16* Wz = W + (size_t)z * wStrideZ;
    const float*    Bz = bias + (size_t)z * biasStrideZ;
    float*          Cz = C + (size_t)z * cStrideZ;

    const int m0 = blockIdx.x * 64 + wave * 16;
    const int n0 = blockIdx.y * 64;
    const int hsel = (lane >> 4) * 8;   // 0 or 8 halves
    const int lid = lane & 15;

    const _Float16* aP = A  + (size_t)(m0 + lid) * D_DIM + hsel;
    const _Float16* bP = Wz + (size_t)(n0 + lid) * D_DIM + hsel;

    v8f acc[4] = { {}, {}, {}, {} };

    for (int k0 = 0; k0 < D_DIM; k0 += 32) {
        v8h alo = *(const v8h*)(aP + k0);
        v8h ahi = *(const v8h*)(aP + k0 + 16);
        v16h a = __builtin_shufflevector(alo, ahi,
                 0, 1, 2, 3, 4, 5, 6, 7, 8, 9, 10, 11, 12, 13, 14, 15);
#pragma unroll
        for (int j = 0; j < 4; ++j) {
            const _Float16* bj = bP + (size_t)j * 16 * D_DIM + k0;
            v8h blo = *(const v8h*)(bj);
            v8h bhi = *(const v8h*)(bj + 16);
            v16h bm = __builtin_shufflevector(blo, bhi,
                      0, 1, 2, 3, 4, 5, 6, 7, 8, 9, 10, 11, 12, 13, 14, 15);
            acc[j] = __builtin_amdgcn_wmma_f32_16x16x32_f16(
                         false, a, false, bm, (short)0, acc[j], false, false);
        }
    }

    // C/D layout: lanes 0-15 -> rows m0..m0+7 (vgpr i = row i), lanes 16-31 -> rows m0+8..15
    const int rowBase = m0 + (lane >> 4) * 8;
#pragma unroll
    for (int j = 0; j < 4; ++j) {
        const int col = n0 + j * 16 + lid;
        const float bv = Bz[col];
#pragma unroll
        for (int i = 0; i < 8; ++i)
            Cz[(size_t)(rowBase + i) * D_DIM + col] = acc[j][i] + bv;
    }
}

// ---------------------------------------------------------------------------
// Relation-aware weighted scatter: one wave32 per edge, all relations one pass
// coef = w_e * inv_c[r][dst] / norm_const[r]
// ---------------------------------------------------------------------------
__global__ __launch_bounds__(256) void scatter_rel_kernel(const float* __restrict__ xr4,
                                                          const int* __restrict__ ei,
                                                          const int* __restrict__ et,
                                                          const float* __restrict__ w_e,
                                                          const float* __restrict__ inv_c,
                                                          const float* __restrict__ nc,
                                                          float* __restrict__ out) {
    const int e = blockIdx.x * 8 + (threadIdx.x >> 5);
    const int lane = threadIdx.x & 31;
    const int src = ei[e];
    const int dst = ei[N_EDGES + e];
    const int r = et[e];
    const float coef = w_e[e] * inv_c[(size_t)r * N_NODES + dst] / nc[r];
    const float* xs = xr4 + ((size_t)r * N_NODES + src) * D_DIM;
    float* od = out + (size_t)dst * D_DIM;
#pragma unroll
    for (int j = 0; j < 12; ++j) {
        int d = lane + j * 32;
        atomicAdd(od + d, coef * xs[d]);
    }
}

// ---------------------------------------------------------------------------
// Plain scatter-mean: aggr[dst] += mj[src] * (1/max(deg,1))
// ---------------------------------------------------------------------------
__global__ __launch_bounds__(256) void scatter_mean_kernel(const float* __restrict__ mj,
                                                           const int* __restrict__ ei,
                                                           const float* __restrict__ inv_all,
                                                           float* __restrict__ aggr) {
    const int e = blockIdx.x * 8 + (threadIdx.x >> 5);
    const int lane = threadIdx.x & 31;
    const int src = ei[e];
    const int dst = ei[N_EDGES + e];
    const float coef = inv_all[dst];
    const float* xs = mj + (size_t)src * D_DIM;
    float* od = aggr + (size_t)dst * D_DIM;
#pragma unroll
    for (int j = 0; j < 12; ++j) {
        int d = lane + j * 32;
        atomicAdd(od + d, coef * xs[d]);
    }
}

// rel_out = relu(out), stored f16 for the next GEMMs
__global__ __launch_bounds__(256) void relu_cvt_kernel(const float* __restrict__ o,
                                                       _Float16* __restrict__ r16, int n) {
    int i = blockIdx.x * 256 + threadIdx.x;
    if (i < n) r16[i] = (_Float16)fmaxf(o[i], 0.0f);
}

// h = relu(aggr + selfout) + h ; refresh h16
__global__ __launch_bounds__(256) void fuse_update_kernel(const float* __restrict__ aggr,
                                                          const float* __restrict__ selfo,
                                                          float* __restrict__ h,
                                                          _Float16* __restrict__ h16, int n) {
    int i = blockIdx.x * 256 + threadIdx.x;
    if (i < n) {
        float v = fmaxf(aggr[i] + selfo[i], 0.0f) + h[i];
        h[i] = v;
        h16[i] = (_Float16)v;
    }
}

// d_out[b][d] = mean over 128 consecutive nodes
__global__ __launch_bounds__(384) void output_mean_kernel(const float* __restrict__ h,
                                                          float* __restrict__ out) {
    const int b = blockIdx.x;
    const int d = threadIdx.x;
    float s = 0.f;
    for (int i = 0; i < 128; ++i)
        s += h[((size_t)(b * 128 + i)) * D_DIM + d];
    out[b * D_DIM + d] = s * (1.0f / 128.0f);
}

// ---------------------------------------------------------------------------
extern "C" void kernel_launch(void* const* d_in, const int* in_sizes, int n_in,
                              void* d_out, int out_size, void* d_ws, size_t ws_size,
                              hipStream_t stream) {
    const float* x        = (const float*)d_in[0];
    const int*   ei       = (const int*)d_in[1];
    const int*   et       = (const int*)d_in[2];
    const float* bn_gamma = (const float*)d_in[3];
    const float* bn_beta  = (const float*)d_in[4];
    const float* W_rel    = (const float*)d_in[5];
    const float* b_rel    = (const float*)d_in[6];
    const float* norm_c   = (const float*)d_in[7];
    const float* W_msg    = (const float*)d_in[8];
    const float* b_msg    = (const float*)d_in[9];
    const float* W_self   = (const float*)d_in[10];
    const float* b_self   = (const float*)d_in[11];
    float* outp = (float*)d_out;

    char* p = (char*)d_ws;
    auto carve = [&](size_t bytes) -> char* {
        char* q = p;
        p += (bytes + 255) & ~(size_t)255;
        return q;
    };

    float*    h      = (float*)carve(ND * 4);
    _Float16* h16    = (_Float16*)carve(ND * 2);
    float*    rnorm  = (float*)carve((size_t)N_NODES * 4);
    float*    w_e    = (float*)carve((size_t)N_EDGES * 4);
    float*    mu     = (float*)carve((size_t)D_DIM * 4);
    float*    rs     = (float*)carve((size_t)D_DIM * 4);
    _Float16* Wrel16 = (_Float16*)carve((size_t)N_LAYERS * N_REL * D_DIM * D_DIM * 2);
    _Float16* Wmsg16 = (_Float16*)carve((size_t)N_LAYERS * D_DIM * D_DIM * 2);
    _Float16* Wslf16 = (_Float16*)carve((size_t)N_LAYERS * D_DIM * D_DIM * 2);
    float*    invc   = (float*)carve((size_t)(N_REL + 1) * N_NODES * 4);  // [4*N] rel, [N] all
    float*    invall = invc + (size_t)N_REL * N_NODES;
    float*    xr4    = (float*)carve((size_t)N_REL * ND * 4);   // also reused: mj/self/aggr
    float*    outb   = (float*)carve(ND * 4);
    _Float16* rel16  = (_Float16*)carve(ND * 2);

    float* mj    = xr4;            // slot 0 (xr dead after scatter_rel)
    float* selfo = xr4 + ND;       // slot 1
    float* aggr  = xr4 + 2 * ND;   // slot 2

    const int DD = D_DIM * D_DIM;

    // ---- preamble -------------------------------------------------------
    bn_stats_kernel<<<D_DIM, 256, 0, stream>>>(x, mu, rs);
    bn_apply_kernel<<<N_NODES, 128, 0, stream>>>(x, bn_gamma, bn_beta, mu, rs, h, h16, rnorm);
    edge_weight_kernel<<<N_EDGES / 8, 256, 0, stream>>>(h, rnorm, ei, w_e);

    cvt_f32_f16_kernel<<<(N_LAYERS * N_REL * DD + 255) / 256, 256, 0, stream>>>(W_rel, Wrel16, N_LAYERS * N_REL * DD);
    cvt_f32_f16_kernel<<<(N_LAYERS * DD + 255) / 256, 256, 0, stream>>>(W_msg, Wmsg16, N_LAYERS * DD);
    cvt_f32_f16_kernel<<<(N_LAYERS * DD + 255) / 256, 256, 0, stream>>>(W_self, Wslf16, N_LAYERS * DD);

    fill_zero_kernel<<<((N_REL + 1) * N_NODES + 255) / 256, 256, 0, stream>>>(invc, (N_REL + 1) * N_NODES);
    count_kernel<<<N_EDGES / 256, 256, 0, stream>>>(ei, et, invc, invall);
    inv_clamp_kernel<<<((N_REL + 1) * N_NODES + 255) / 256, 256, 0, stream>>>(invc, (N_REL + 1) * N_NODES);

    const dim3 gemmBlk(128);
    const dim3 gemmGridR(N_NODES / 64, D_DIM / 64, N_REL);
    const dim3 gemmGrid1(N_NODES / 64, D_DIM / 64, 1);
    const int  ndInt = (int)ND;

    // ---- layers ---------------------------------------------------------
    for (int l = 0; l < N_LAYERS; ++l) {
        // per-relation linear: xr4[r] = h @ W_rel[l,r].T + b_rel[l,r]
        gemm_f16_wmma_kernel<<<gemmGridR, gemmBlk, 0, stream>>>(
            h16, Wrel16 + (size_t)l * N_REL * DD, b_rel + (size_t)l * N_REL * D_DIM, xr4,
            (long)DD, (long)D_DIM, (long)ND);

        fill_zero_kernel<<<(ndInt + 255) / 256, 256, 0, stream>>>(outb, ndInt);
        scatter_rel_kernel<<<N_EDGES / 8, 256, 0, stream>>>(
            xr4, ei, et, w_e, invc, norm_c + (size_t)l * N_REL, outb);

        relu_cvt_kernel<<<(ndInt + 255) / 256, 256, 0, stream>>>(outb, rel16, ndInt);

        // mj = rel_out @ W_msg.T + b_msg ; selfo = rel_out @ W_self.T + b_self
        gemm_f16_wmma_kernel<<<gemmGrid1, gemmBlk, 0, stream>>>(
            rel16, Wmsg16 + (size_t)l * DD, b_msg + (size_t)l * D_DIM, mj, 0L, 0L, 0L);
        gemm_f16_wmma_kernel<<<gemmGrid1, gemmBlk, 0, stream>>>(
            rel16, Wslf16 + (size_t)l * DD, b_self + (size_t)l * D_DIM, selfo, 0L, 0L, 0L);

        fill_zero_kernel<<<(ndInt + 255) / 256, 256, 0, stream>>>(aggr, ndInt);
        scatter_mean_kernel<<<N_EDGES / 8, 256, 0, stream>>>(mj, ei, invall, aggr);

        fuse_update_kernel<<<(ndInt + 255) / 256, 256, 0, stream>>>(aggr, selfo, h, h16, ndInt);
    }

    // ---- output ---------------------------------------------------------
    output_mean_kernel<<<BATCH, D_DIM, 0, stream>>>(h, outp);
}
